// RCNN_65549790872110
// MI455X (gfx1250) — matrix operand order repaired and checked by reading, not compiled, not run on hardware
//
#include <hip/hip_runtime.h>
#include <math.h>

// ---------------------------------------------------------------------------
// RCNN tree-RNN for MI455X (gfx1250): level-parallel bottom-up evaluation.
//   - leaves  (2048..16383): out = tanh(inp @ W[:, :512].T + b)   [C==1 pool]
//   - internal (5 levels):   [8,1024] x [1024,256] per node, tanh, reshape-pool
// Matrix cores: v_wmma_f32_16x16x32_bf16, wave32, 8 waves/block.
// ---------------------------------------------------------------------------

typedef __attribute__((ext_vector_type(16))) __bf16        v16bf;
typedef __attribute__((ext_vector_type(8)))  float         v8f;
typedef __attribute__((ext_vector_type(8)))  unsigned int  v8u;

#define NNODES   16384
#define MAXCH    8
#define IN_DIM   512
#define MEM_DIM  256
#define FAN_IN   1024           // IN_DIM + 2*MEM_DIM
#define LEAF_LO  2048           // first node with zero children

__device__ __forceinline__ unsigned short f2bf(float f) {
    // round-to-nearest-even f32 -> bf16 (NaN payload not preserved; fine here)
    unsigned int u = __float_as_uint(f);
    u += 0x7FFFu + ((u >> 16) & 1u);
    return (unsigned short)(u >> 16);
}

__device__ __forceinline__ v16bf frag_from(uint4 lo, uint4 hi) {
    v8u p;
    p[0] = lo.x; p[1] = lo.y; p[2] = lo.z; p[3] = lo.w;
    p[4] = hi.x; p[5] = hi.y; p[6] = hi.z; p[7] = hi.w;
    return __builtin_bit_cast(v16bf, p);
}

__device__ __forceinline__ v16bf frag_from8(v8u p) {
    return __builtin_bit_cast(v16bf, p);
}

// --------------------------- W f32 -> bf16 ---------------------------------
__global__ void cvt_w_kernel(const float* __restrict__ W,
                             unsigned short* __restrict__ Wbf) {
    int i = blockIdx.x * blockDim.x + threadIdx.x;
    int stride = gridDim.x * blockDim.x;
    for (; i < MEM_DIM * FAN_IN; i += stride) Wbf[i] = f2bf(W[i]);
}

// ------------------------------- leaves ------------------------------------
// 16 leaves per block; A = [16 x 512] bf16 in LDS; 8 waves x 2 N-tiles.
__global__ __launch_bounds__(256, 1)
void leaf_kernel(const float* __restrict__ inputs,
                 const unsigned short* __restrict__ Wbf,
                 const float* __restrict__ bias,
                 float* __restrict__ out,
                 unsigned short* __restrict__ Sbf) {
    __shared__ __align__(16) unsigned short Als[16][IN_DIM];

    const int t = threadIdx.x;
    const int tile_base = LEAF_LO + blockIdx.x * 16;

    // Build A tile: each thread converts 32 contiguous floats of one row.
    {
        const int row = t >> 4;
        const int colbase = (t & 15) * 32;
        const float* src = inputs + (size_t)(tile_base + row) * IN_DIM + colbase;
#pragma unroll
        for (int i = 0; i < 8; ++i) {
            float4 v = *(const float4*)(src + i * 4);
            unsigned int w0 = (unsigned)f2bf(v.x) | ((unsigned)f2bf(v.y) << 16);
            unsigned int w1 = (unsigned)f2bf(v.z) | ((unsigned)f2bf(v.w) << 16);
            *(unsigned int*)&Als[row][colbase + i * 4]     = w0;
            *(unsigned int*)&Als[row][colbase + i * 4 + 2] = w1;
        }
    }
    __syncthreads();

    const int lane  = t & 31;
    const int wave  = t >> 5;
    const int nloc  = lane & 15;   // A row (M) / B column (N) owned by lane
    const int khalf = lane >> 4;
    const int nt0 = wave * 2, nt1 = wave * 2 + 1;

    const unsigned short* Wr0 = Wbf + (size_t)(nt0 * 16 + nloc) * FAN_IN;
    const unsigned short* Wr1 = Wbf + (size_t)(nt1 * 16 + nloc) * FAN_IN;

    v8f acc0 = {}; v8f acc1 = {};
    for (int kc = 0; kc < IN_DIM / 32; ++kc) {
        const int kbase = kc * 32;
        // A frag: K = {kh*16..+15} of row nloc
        uint4 alo = *(const uint4*)&Als[nloc][kbase + khalf * 16];
        uint4 ahi = *(const uint4*)&Als[nloc][kbase + khalf * 16 + 8];
        v16bf a = frag_from(alo, ahi);
        // B frag: 16 contiguous K of W row (= B column), per half-wave
        v8u b0 = *(const v8u*)(Wr0 + kbase + khalf * 16);
        v8u b1 = *(const v8u*)(Wr1 + kbase + khalf * 16);
        acc0 = __builtin_amdgcn_wmma_f32_16x16x32_bf16(false, a, false, frag_from8(b0),
                                                       (short)0, acc0, false, false);
        acc1 = __builtin_amdgcn_wmma_f32_16x16x32_bf16(false, a, false, frag_from8(b1),
                                                       (short)0, acc1, false, false);
    }

    // Epilogue: C layout — VGPR v: (lane<16) -> M=v, else M=v+8; N = lane&15.
    const int mhi = khalf * 8;
    const int f0 = nt0 * 16 + nloc;
    const int f1 = nt1 * 16 + nloc;
    const float bb0 = bias[f0], bb1 = bias[f1];
#pragma unroll
    for (int v = 0; v < 8; ++v) {
        const int node = tile_base + v + mhi;
        float r0 = tanhf(acc0[v] + bb0);
        float r1 = tanhf(acc1[v] + bb1);
        out[(size_t)node * MEM_DIM + f0] = r0;
        out[(size_t)node * MEM_DIM + f1] = r1;
        Sbf[(size_t)node * MEM_DIM + f0] = f2bf(r0);
        Sbf[(size_t)node * MEM_DIM + f1] = f2bf(r1);
    }
}

// --------------------------- internal levels -------------------------------
// 2 nodes per block (16 A rows = 2 nodes x 8 child slots), K = 1024.
__global__ __launch_bounds__(256, 1)
void inner_kernel(const float* __restrict__ inputs,
                  const unsigned short* __restrict__ Wbf,
                  const float* __restrict__ bias,
                  const float* __restrict__ emb,
                  const int* __restrict__ child_idx,
                  const int* __restrict__ child_deprel,
                  const int* __restrict__ counts,
                  float* __restrict__ out,
                  unsigned short* __restrict__ Sbf,
                  int lo, int hi) {
    __shared__ __align__(16) unsigned short Als[16][FAN_IN];
    __shared__ float Rs[16][MEM_DIM];

    const int t = threadIdx.x;
    const int base = lo + blockIdx.x * 2;

    // ---- build A tile: row = node_local*8 + slot; cols [inp | dep | child]
    {
        const int row = t >> 4;
        const int colbase = (t & 15) * 64;       // 64 cols per thread, one segment
        int node = base + (row >> 3);
        if (node >= hi) node = hi - 1;           // duplicate last node; stores guarded
        const int s = row & 7;
        const int cidx = child_idx[node * MAXCH + s];
        const bool real = (cidx >= 0);

        if (colbase < IN_DIM) {                  // broadcast input segment
            const float* src = inputs + (size_t)node * IN_DIM + colbase;
#pragma unroll
            for (int i = 0; i < 16; ++i) {
                float4 v = *(const float4*)(src + i * 4);
                unsigned int w0 = (unsigned)f2bf(v.x) | ((unsigned)f2bf(v.y) << 16);
                unsigned int w1 = (unsigned)f2bf(v.z) | ((unsigned)f2bf(v.w) << 16);
                *(unsigned int*)&Als[row][colbase + i * 4]     = w0;
                *(unsigned int*)&Als[row][colbase + i * 4 + 2] = w1;
            }
        } else if (colbase < IN_DIM + MEM_DIM) { // deprel embedding segment
            const int c = colbase - IN_DIM;
            if (real) {
                const int d = child_deprel[node * MAXCH + s];
                const float* src = emb + (size_t)d * MEM_DIM + c;
#pragma unroll
                for (int i = 0; i < 16; ++i) {
                    float4 v = *(const float4*)(src + i * 4);
                    unsigned int w0 = (unsigned)f2bf(v.x) | ((unsigned)f2bf(v.y) << 16);
                    unsigned int w1 = (unsigned)f2bf(v.z) | ((unsigned)f2bf(v.w) << 16);
                    *(unsigned int*)&Als[row][colbase + i * 4]     = w0;
                    *(unsigned int*)&Als[row][colbase + i * 4 + 2] = w1;
                }
            } else {
                uint4 z = {0u, 0u, 0u, 0u};
#pragma unroll
                for (int i = 0; i < 8; ++i)
                    *(uint4*)&Als[row][colbase + i * 8] = z;
            }
        } else {                                 // child state segment (already bf16)
            const int c = colbase - (IN_DIM + MEM_DIM);
            if (real) {
                const unsigned short* src = Sbf + (size_t)cidx * MEM_DIM + c;
#pragma unroll
                for (int i = 0; i < 8; ++i)
                    *(uint4*)&Als[row][colbase + i * 8] = *(const uint4*)(src + i * 8);
            } else {
                uint4 z = {0u, 0u, 0u, 0u};
#pragma unroll
                for (int i = 0; i < 8; ++i)
                    *(uint4*)&Als[row][colbase + i * 8] = z;
            }
        }
    }
    __syncthreads();

    // ---- GEMM: 32 K-chunks of 32, 2 N-tiles per wave
    const int lane  = t & 31;
    const int wave  = t >> 5;
    const int nloc  = lane & 15;
    const int khalf = lane >> 4;
    const int nt0 = wave * 2, nt1 = wave * 2 + 1;
    const unsigned short* Wr0 = Wbf + (size_t)(nt0 * 16 + nloc) * FAN_IN;
    const unsigned short* Wr1 = Wbf + (size_t)(nt1 * 16 + nloc) * FAN_IN;

    v8f acc0 = {}; v8f acc1 = {};
    for (int kc = 0; kc < FAN_IN / 32; ++kc) {
        const int kbase = kc * 32;
        uint4 alo = *(const uint4*)&Als[nloc][kbase + khalf * 16];
        uint4 ahi = *(const uint4*)&Als[nloc][kbase + khalf * 16 + 8];
        v16bf a = frag_from(alo, ahi);
        v8u b0 = *(const v8u*)(Wr0 + kbase + khalf * 16);
        v8u b1 = *(const v8u*)(Wr1 + kbase + khalf * 16);
        acc0 = __builtin_amdgcn_wmma_f32_16x16x32_bf16(false, a, false, frag_from8(b0),
                                                       (short)0, acc0, false, false);
        acc1 = __builtin_amdgcn_wmma_f32_16x16x32_bf16(false, a, false, frag_from8(b1),
                                                       (short)0, acc1, false, false);
    }

    // ---- stage r = tanh(acc + b) into LDS for the reshape-pool
    {
        const int mhi = khalf * 8;
        const int f0 = nt0 * 16 + nloc;
        const int f1 = nt1 * 16 + nloc;
        const float bb0 = bias[f0], bb1 = bias[f1];
#pragma unroll
        for (int v = 0; v < 8; ++v) {
            Rs[v + mhi][f0] = tanhf(acc0[v] + bb0);
            Rs[v + mhi][f1] = tanhf(acc1[v] + bb1);
        }
    }
    __syncthreads();

    // ---- pooling: out[f] = max_{tt<C} r_flat[f*C + tt], r_flat row-major [C,256]
    const float NEG = -3.4028234663852886e38f;
#pragma unroll
    for (int u = 0; u < 2; ++u) {
        const int node = base + u;
        if (node >= hi) continue;
        int C = counts[node];
        if (C < 1) C = 1;
        const int f = t;                       // one feature per thread
        float m = NEG;
        for (int tt = 0; tt < C; ++tt) {
            const int q = f * C + tt;          // q/256 <= C-1 < 8
            m = fmaxf(m, Rs[u * 8 + (q >> 8)][q & 255]);
        }
        out[(size_t)node * MEM_DIM + f] = m;
        Sbf[(size_t)node * MEM_DIM + f] = f2bf(m);
    }
}

// ------------------------------- launcher ----------------------------------
extern "C" void kernel_launch(void* const* d_in, const int* in_sizes, int n_in,
                              void* d_out, int out_size, void* d_ws, size_t ws_size,
                              hipStream_t stream) {
    const float* inputs       = (const float*)d_in[0];
    const float* W            = (const float*)d_in[1];
    const float* bias         = (const float*)d_in[2];
    const float* emb          = (const float*)d_in[3];
    const int*   child_idx    = (const int*)d_in[4];
    const int*   child_deprel = (const int*)d_in[5];
    const int*   counts       = (const int*)d_in[6];
    float* out = (float*)d_out;

    // workspace: [ W bf16 : 512KB ][ states bf16 : 8MB ]
    unsigned short* Wbf = (unsigned short*)d_ws;
    unsigned short* Sbf = (unsigned short*)((char*)d_ws + (size_t)MEM_DIM * FAN_IN * 2);

    cvt_w_kernel<<<512, 256, 0, stream>>>(W, Wbf);

    // all 14336 leaves in one batched GEMM pass
    leaf_kernel<<<(NNODES - LEAF_LO) / 16, 256, 0, stream>>>(inputs, Wbf, bias, out, Sbf);

    // internal levels, deepest first (stream order provides the dependency)
    struct { int lo, hi; } lv[5] = { {585, 2048}, {73, 585}, {9, 73}, {1, 9}, {0, 1} };
    for (int i = 0; i < 5; ++i) {
        const int n = lv[i].hi - lv[i].lo;
        inner_kernel<<<(n + 1) / 2, 256, 0, stream>>>(
            inputs, Wbf, bias, emb, child_idx, child_deprel, counts,
            out, Sbf, lv[i].lo, lv[i].hi);
    }
}
